// MultiScaleDilateAttention_28226525069830
// MI455X (gfx1250) — compile-verified, matrix operand order checked by reading, not run on hardware
//
#include <hip/hip_runtime.h>
#include <hip/hip_bf16.h>

typedef __attribute__((ext_vector_type(16))) _Float16 v16h;
typedef __attribute__((ext_vector_type(8)))  float    v8f;

// Problem constants (from reference)
static constexpr int BATCH = 2;
static constexpr int HH    = 128;
static constexpr int WW    = 128;
static constexpr int CC    = 192;          // channels
static constexpr int C3    = 3 * CC;       // 576
static constexpr int MROWS = BATCH * HH * WW;  // 32768
static constexpr int ND    = 3;            // dilations 1,2,3
static constexpr int HEADD = 32;           // head_dim
static constexpr int GC    = CC / ND;      // 64 channels per dilation group
static constexpr int NH_G  = GC / HEADD;   // 2 heads per group
static constexpr int NB    = 4;            // N-tiles (of 16) per wave -> 16x64 strip

// ---------------- prep kernels ----------------

__global__ __launch_bounds__(256) void cvt_f32_to_f16(const float* __restrict__ src,
                                                      _Float16* __restrict__ dst, int n) {
    int i = blockIdx.x * blockDim.x + threadIdx.x;
    int stride = gridDim.x * blockDim.x;
    for (; i < n; i += stride) dst[i] = (_Float16)src[i];
}

// W [K,N] f32 row-major  ->  Wt [N,K] f16 row-major (transposed)
__global__ __launch_bounds__(256) void transpose_cvt(const float* __restrict__ W,
                                                     _Float16* __restrict__ Wt,
                                                     int K, int N) {
    int i = blockIdx.x * blockDim.x + threadIdx.x;
    int total = K * N;
    int stride = gridDim.x * blockDim.x;
    for (; i < total; i += stride) {
        int n = i / K;
        int k = i - n * K;
        Wt[i] = (_Float16)W[(size_t)k * N + n];
    }
}

// ---------------- WMMA GEMM ----------------
// A  : [M,K] f16 row-major
// Bt : [N,K] f16 row-major (i.e. B transposed)
// One wave computes a 16(M) x 64(N) strip: 1 A fragment + 4 B fragments + 4 WMMA
// per 32-wide k-step.
// Per-lane fragment layout (ISA 7.12.2, 16-bit A 16x32):
//   lane<16 : row M=lane,    halves[0..7]=K0..7,  halves[8..15]=K16..23
//   lane>=16: row M=lane-16, halves[0..7]=K8..15, halves[8..15]=K24..31
// B mirrored with N in place of M (one column of B per lane).

union FragU { v16h v; unsigned int u[8]; };

__device__ __forceinline__ void load_frag(const _Float16* __restrict__ base,
                                          int row, int ldk, int k0,
                                          int lhi, FragU& f) {
    const unsigned int* p =
        (const unsigned int*)(base + (size_t)(row) * ldk + k0 + 8 * lhi);
#pragma unroll
    for (int j = 0; j < 4; ++j) f.u[j] = p[j];
#pragma unroll
    for (int j = 0; j < 4; ++j) f.u[4 + j] = p[8 + j];
}

__global__ __launch_bounds__(256) void gemm_wmma_f16out(const _Float16* __restrict__ A,
                                                        const _Float16* __restrict__ Bt,
                                                        _Float16* __restrict__ C,
                                                        int M, int N, int K) {
    int wave = blockIdx.x * (blockDim.x >> 5) + (threadIdx.x >> 5);
    int stripsN = N >> 6;                       // strips of 64 columns
    int totalStrips = (M >> 4) * stripsN;
    if (wave >= totalStrips) return;
    int tm = (wave / stripsN) << 4;
    int tn = (wave % stripsN) << 6;
    int lane = threadIdx.x & 31;
    int lrow = lane & 15;
    int lhi  = lane >> 4;

    v8f acc[NB];
#pragma unroll
    for (int q = 0; q < NB; ++q)
        acc[q] = (v8f){0.f, 0.f, 0.f, 0.f, 0.f, 0.f, 0.f, 0.f};

    for (int k0 = 0; k0 < K; k0 += 32) {
        FragU fa;
        load_frag(A, tm + lrow, K, k0, lhi, fa);
        FragU fb[NB];
#pragma unroll
        for (int q = 0; q < NB; ++q)
            load_frag(Bt, tn + q * 16 + lrow, K, k0, lhi, fb[q]);
#pragma unroll
        for (int q = 0; q < NB; ++q)
            acc[q] = __builtin_amdgcn_wmma_f32_16x16x32_f16(
                false, fa.v, false, fb[q].v, (short)0, acc[q], false, false);
    }
    int mbase = tm + lhi * 8;
#pragma unroll
    for (int q = 0; q < NB; ++q) {
        int n = tn + q * 16 + lrow;
#pragma unroll
        for (int j = 0; j < 8; ++j)
            C[(size_t)(mbase + j) * N + n] = (_Float16)acc[q][j];
    }
}

__global__ __launch_bounds__(256) void gemm_wmma_f32bias(const _Float16* __restrict__ A,
                                                         const _Float16* __restrict__ Bt,
                                                         const float* __restrict__ bias,
                                                         float* __restrict__ C,
                                                         int M, int N, int K) {
    int wave = blockIdx.x * (blockDim.x >> 5) + (threadIdx.x >> 5);
    int stripsN = N >> 6;
    int totalStrips = (M >> 4) * stripsN;
    if (wave >= totalStrips) return;
    int tm = (wave / stripsN) << 4;
    int tn = (wave % stripsN) << 6;
    int lane = threadIdx.x & 31;
    int lrow = lane & 15;
    int lhi  = lane >> 4;

    v8f acc[NB];
#pragma unroll
    for (int q = 0; q < NB; ++q)
        acc[q] = (v8f){0.f, 0.f, 0.f, 0.f, 0.f, 0.f, 0.f, 0.f};

    for (int k0 = 0; k0 < K; k0 += 32) {
        FragU fa;
        load_frag(A, tm + lrow, K, k0, lhi, fa);
        FragU fb[NB];
#pragma unroll
        for (int q = 0; q < NB; ++q)
            load_frag(Bt, tn + q * 16 + lrow, K, k0, lhi, fb[q]);
#pragma unroll
        for (int q = 0; q < NB; ++q)
            acc[q] = __builtin_amdgcn_wmma_f32_16x16x32_f16(
                false, fa.v, false, fb[q].v, (short)0, acc[q], false, false);
    }
    int mbase = tm + lhi * 8;
#pragma unroll
    for (int q = 0; q < NB; ++q) {
        int n = tn + q * 16 + lrow;
        float b = bias[n];
#pragma unroll
        for (int j = 0; j < 8; ++j)
            C[(size_t)(mbase + j) * N + n] = acc[q][j] + b;
    }
}

// ---------------- dilated attention ----------------
// q from q_qkv (slot 0), k/v from kv_qkv (slots 1,2).
// qkv row stride = 576; column = slot*192 + dil_i*64 + head*32 + ch.
// out row stride = 192; column = dil_i*64 + head*32 + ch.
__global__ __launch_bounds__(256) void dilate_attn(const _Float16* __restrict__ q_qkv,
                                                   const _Float16* __restrict__ kv_qkv,
                                                   _Float16* __restrict__ outbuf) {
    const int items = MROWS * ND * NH_G;  // 196608
    int idx = blockIdx.x * blockDim.x + threadIdx.x;
    if (idx >= items) return;

    int p    = idx % MROWS;               // pixel index (b*H+y)*W+x
    int dh   = idx / MROWS;
    int dili = dh >> 1;                   // 0..2
    int head = dh & 1;                    // 0..1
    const int dil = dili + 1;             // dilations (1,2,3)

    int b  = p / (HH * WW);
    int yy = (p / WW) % HH;
    int xx = p % WW;

    const float scale = 0.17677669529663687f;  // 32^-0.5

    const int grp = dili * GC + head * HEADD;
    const _Float16* qptr = q_qkv + (size_t)p * C3 + grp;

    float q[HEADD];
#pragma unroll
    for (int c = 0; c < HEADD; ++c) q[c] = (float)qptr[c];

    float logits[9];
    int   p2s[9];
    float mx = -1e30f;
#pragma unroll
    for (int a = 0; a < 3; ++a) {
#pragma unroll
        for (int bb = 0; bb < 3; ++bb) {
            int t  = a * 3 + bb;
            int y2 = yy + (a - 1) * dil;
            int x2 = xx + (bb - 1) * dil;
            float lg = 0.f;       // zero-padded k => dot = 0 (matches reference)
            int p2 = -1;
            if (y2 >= 0 && y2 < HH && x2 >= 0 && x2 < WW) {
                p2 = (b * HH + y2) * WW + x2;
                const _Float16* kptr = kv_qkv + (size_t)p2 * C3 + CC + grp;
                float d = 0.f;
#pragma unroll
                for (int c = 0; c < HEADD; ++c) d += q[c] * (float)kptr[c];
                lg = d * scale;
            }
            logits[t] = lg;
            p2s[t] = p2;
            mx = fmaxf(mx, lg);
        }
    }

    float den = 0.f;
#pragma unroll
    for (int t = 0; t < 9; ++t) {
        logits[t] = __expf(logits[t] - mx);
        den += logits[t];
    }
    float inv = 1.f / den;

    float acc[HEADD];
#pragma unroll
    for (int c = 0; c < HEADD; ++c) acc[c] = 0.f;
#pragma unroll
    for (int t = 0; t < 9; ++t) {
        if (p2s[t] < 0) continue;   // zero-padded v contributes nothing
        float w = logits[t] * inv;
        const _Float16* vptr = kv_qkv + (size_t)p2s[t] * C3 + 2 * CC + grp;
#pragma unroll
        for (int c = 0; c < HEADD; ++c) acc[c] += w * (float)vptr[c];
    }

    _Float16* optr = outbuf + (size_t)p * CC + grp;
#pragma unroll
    for (int c = 0; c < HEADD; ++c) optr[c] = (_Float16)acc[c];
}

// ---------------- launch ----------------

extern "C" void kernel_launch(void* const* d_in, const int* in_sizes, int n_in,
                              void* d_out, int out_size, void* d_ws, size_t ws_size,
                              hipStream_t stream) {
    const float* mov     = (const float*)d_in[0];
    const float* fix     = (const float*)d_in[1];
    const float* W_mqkv  = (const float*)d_in[2];
    const float* W_fqkv  = (const float*)d_in[3];
    const float* W_mproj = (const float*)d_in[4];
    const float* b_mproj = (const float*)d_in[5];
    const float* W_fproj = (const float*)d_in[6];
    const float* b_fproj = (const float*)d_in[7];
    float* out = (float*)d_out;

    // workspace carving (halves)
    _Float16* ws = (_Float16*)d_ws;
    const size_t SZ_ACT  = (size_t)MROWS * CC;   // 6291456
    const size_t SZ_WQKV = (size_t)C3 * CC;      // 110592
    const size_t SZ_WPRJ = (size_t)CC * CC;      // 36864
    const size_t SZ_QKV  = (size_t)MROWS * C3;   // 18874368

    _Float16* movh    = ws;                    size_t o = SZ_ACT;
    _Float16* fixh    = ws + o;                o += SZ_ACT;
    _Float16* Wm_t    = ws + o;                o += SZ_WQKV;
    _Float16* Wf_t    = ws + o;                o += SZ_WQKV;
    _Float16* Wmp_t   = ws + o;                o += SZ_WPRJ;
    _Float16* Wfp_t   = ws + o;                o += SZ_WPRJ;
    _Float16* movqkv  = ws + o;                o += SZ_QKV;
    _Float16* fixqkv  = ws + o;                o += SZ_QKV;
    _Float16* movattn = ws + o;                o += SZ_ACT;
    _Float16* fixattn = ws + o;                o += SZ_ACT;

    // 1) convert activations to f16
    {
        int n = (int)SZ_ACT;
        int blocks = (n + 255) / 256;
        if (blocks > 4096) blocks = 4096;
        cvt_f32_to_f16<<<blocks, 256, 0, stream>>>(mov, movh, n);
        cvt_f32_to_f16<<<blocks, 256, 0, stream>>>(fix, fixh, n);
    }
    // 2) transpose+convert weights
    {
        int n1 = CC * C3, b1 = (n1 + 255) / 256;
        transpose_cvt<<<b1, 256, 0, stream>>>(W_mqkv, Wm_t, CC, C3);
        transpose_cvt<<<b1, 256, 0, stream>>>(W_fqkv, Wf_t, CC, C3);
        int n2 = CC * CC, b2 = (n2 + 255) / 256;
        transpose_cvt<<<b2, 256, 0, stream>>>(W_mproj, Wmp_t, CC, CC);
        transpose_cvt<<<b2, 256, 0, stream>>>(W_fproj, Wfp_t, CC, CC);
    }
    // 3) QKV GEMMs (WMMA):  [32768,192] x [192,576], 16x64 strip per wave
    {
        int strips = (MROWS / 16) * (C3 / 64);      // 18432
        int blocks = (strips + 7) / 8;              // 8 waves per block
        gemm_wmma_f16out<<<blocks, 256, 0, stream>>>(movh, Wm_t, movqkv, MROWS, C3, CC);
        gemm_wmma_f16out<<<blocks, 256, 0, stream>>>(fixh, Wf_t, fixqkv, MROWS, C3, CC);
    }
    // 4) dilated cross attention
    {
        int items = MROWS * ND * NH_G;              // 196608
        int blocks = (items + 255) / 256;
        // mov branch: q = fix, k/v = mov
        dilate_attn<<<blocks, 256, 0, stream>>>(fixqkv, movqkv, movattn);
        // fix branch: q = mov, k/v = fix
        dilate_attn<<<blocks, 256, 0, stream>>>(movqkv, fixqkv, fixattn);
    }
    // 5) output projections (WMMA) + bias, write f32 to d_out (mov | fix)
    {
        int strips = (MROWS / 16) * (CC / 64);      // 6144
        int blocks = (strips + 7) / 8;
        gemm_wmma_f32bias<<<blocks, 256, 0, stream>>>(movattn, Wmp_t, b_mproj,
                                                      out, MROWS, CC, CC);
        gemm_wmma_f32bias<<<blocks, 256, 0, stream>>>(fixattn, Wfp_t, b_fproj,
                                                      out + (size_t)MROWS * CC, MROWS, CC, CC);
    }
}
